// Quantisation_60902636257595
// MI455X (gfx1250) — matrix operand order repaired and checked by reading, not compiled, not run on hardware
//
#include <hip/hip_runtime.h>

// ---- CDNA5 WMMA vector types (wave32) ----
typedef __attribute__((ext_vector_type(16))) __bf16 v16bf;
typedef __attribute__((ext_vector_type(8)))  float  v8f;

// f32 -> bf16, round-to-nearest-even
static __device__ __forceinline__ __bf16 f2bf(float f) {
    union { float f; unsigned int u; } v; v.f = f;
    unsigned int r = v.u + 0x7FFFu + ((v.u >> 16) & 1u);
    unsigned short h = (unsigned short)(r >> 16);
    return __builtin_bit_cast(__bf16, h);
}

// LDS byte offset of a __shared__ object (generic LDS pointers carry the
// wave-relative offset in the low 32 bits).
static __device__ __forceinline__ unsigned lds_off(const void* p) {
    return (unsigned)(unsigned long long)p;
}

// Async DMA copy of one 8KB code tile (16 codes x 256 ch bf16) global -> LDS.
// 256 threads x 2 x 16B via GLOBAL_LOAD_ASYNC_TO_LDS_B128 (ASYNCcnt path);
// INST_OFFSET is added to BOTH the LDS and global addresses (ISA 08 §4.4).
static __device__ __forceinline__ void async_tile_copy(unsigned lds_base,
                                                       const unsigned short* gbase,
                                                       unsigned byte_off, int tid) {
    unsigned l = lds_base + (unsigned)tid * 16u;
    unsigned g = byte_off + (unsigned)tid * 16u;
    asm volatile("global_load_async_to_lds_b128 %0, %1, %2"
                 :: "v"(l), "v"(g), "s"(gbase) : "memory");
    asm volatile("global_load_async_to_lds_b128 %0, %1, %2 offset:4096"
                 :: "v"(l), "v"(g), "s"(gbase) : "memory");
}

static __device__ __forceinline__ void wait_async0() {
    asm volatile("s_wait_asynccnt 0x0" ::: "memory");
}

// -------- Prep: codebook f32 -> bf16, plus |e_k|^2 --------
__global__ __launch_bounds__(256) void vq_prep(const float* __restrict__ cb,
                                               unsigned short* __restrict__ cbbf_u16,
                                               float* __restrict__ esq) {
    __shared__ float red[256];
    const int k = blockIdx.x;
    const int t = threadIdx.x;
    float v = cb[k * 256 + t];
    __bf16 bv = f2bf(v);
    cbbf_u16[k * 256 + t] = __builtin_bit_cast(unsigned short, bv);
    red[t] = v * v;
    __syncthreads();
    for (int s = 128; s > 0; s >>= 1) {
        if (t < s) red[t] += red[t + s];
        __syncthreads();
    }
    if (t == 0) esq[k] = red[0];
}

// -------- Main: fused dots (WMMA bf16) + argmin + gather --------
// Grid: 512 blocks x 256 threads (8 waves). Block owns 256 pixels of one image.
// Wave w owns pixels [w*32, w*32+32): two 16x16 WMMA row-panels.
__global__ __launch_bounds__(256) void vq_main(const float* __restrict__ x,
                                               const float* __restrict__ cb,
                                               const unsigned short* __restrict__ cbbf_u16,
                                               const float* __restrict__ esq,
                                               float* __restrict__ out) {
    __shared__ __attribute__((aligned(32))) __bf16 btile[2][16 * 256]; // 2 x 8KB code tiles
    __shared__ float esq_lds[1024];
    __shared__ int   idx_lds[256];

    const int tid  = threadIdx.x;
    const int lane = tid & 31;
    const int w    = tid >> 5;
    const int lo   = lane & 15;   // N column / A row within panel
    const int hi   = lane >> 4;   // half-wave selector
    const int b    = blockIdx.x >> 4;
    const int hw0  = (blockIdx.x & 15) << 8;

    const unsigned bt0 = lds_off(&btile[0][0]);
    const unsigned bt1 = bt0 + 8192u;

    // Stage |e|^2 into LDS
    for (int i = tid; i < 1024; i += 256) esq_lds[i] = esq[i];

    // Kick off async DMA of B tile 0 while we assemble A fragments
    async_tile_copy(bt0, cbbf_u16, 0u, tid);

    // ---- Load A fragments (2 panels x 8 k-steps) straight from global, f32->bf16 ----
    // A 16x32 bf16 layout: lanes 0-15: elems0-15 = K(0..15);
    //                      lanes16-31: elems0-15 = K(16..31).
    v16bf a[2][8];
    const float* xb = x + (size_t)b * 256 * 4096 + hw0;
    #pragma unroll
    for (int t = 0; t < 2; ++t) {
        const int pix = w * 32 + t * 16 + lo;
        #pragma unroll
        for (int kk = 0; kk < 8; ++kk) {
            #pragma unroll
            for (int j = 0; j < 8; ++j) {
                const int ch0 = kk * 32 + hi * 16 + j;
                a[t][kk][j]     = f2bf(xb[(size_t)ch0 * 4096 + pix]);
                a[t][kk][j + 8] = f2bf(xb[(size_t)(ch0 + 8) * 4096 + pix]);
            }
        }
    }

    wait_async0();
    __syncthreads();

    float bd0[8], bd1[8];
    int   bi0[8], bi1[8];
    #pragma unroll
    for (int r = 0; r < 8; ++r) { bd0[r] = bd1[r] = 3.4e38f; bi0[r] = bi1[r] = 0; }

    for (int ct = 0; ct < 64; ++ct) {
        const int cur = ct & 1;
        // Async-prefetch next code tile into the other LDS buffer (no VGPR round trip)
        if (ct + 1 < 64) {
            async_tile_copy(cur ? bt0 : bt1, cbbf_u16,
                            (unsigned)(ct + 1) * 8192u, tid);
        }

        // ---- 16 WMMAs: dots for 32 pixels x 16 codes over d=256 ----
        // B 32x16 bf16 layout: lane n<16: elems = e[code n][cbase..cbase+15];
        //                      lane n+16: elems = e[code n][cbase+16..cbase+31].
        v8f acc0 = {}; v8f acc1 = {};
        const __bf16* bp = &btile[cur][0] + lo * 256 + hi * 16;
        #pragma unroll
        for (int kk = 0; kk < 8; ++kk) {
            v16bf bfrag = *(const v16bf*)(bp + kk * 32);
            acc0 = __builtin_amdgcn_wmma_f32_16x16x32_bf16(
                false, a[0][kk], false, bfrag, (short)0, acc0, false, false);
            acc1 = __builtin_amdgcn_wmma_f32_16x16x32_bf16(
                false, a[1][kk], false, bfrag, (short)0, acc1, false, false);
        }

        // dist = -2*dot + |e|^2  (|x|^2 constant per pixel: dropped for argmin)
        const int code = (ct << 4) + lo;            // C/D layout: code fixed per lane
        const float ev = esq_lds[code];
        #pragma unroll
        for (int r = 0; r < 8; ++r) {
            const float d0 = fmaf(-2.0f, acc0[r], ev);
            if (d0 < bd0[r]) { bd0[r] = d0; bi0[r] = code; }
            const float d1 = fmaf(-2.0f, acc1[r], ev);
            if (d1 < bd1[r]) { bd1[r] = d1; bi1[r] = code; }
        }

        // Own async writes complete, then block-wide barrier: double-buffer safe.
        wait_async0();
        __syncthreads();
    }

    // ---- Butterfly argmin across the 16 lanes of each half-wave ----
    #pragma unroll
    for (int t = 0; t < 2; ++t) {
        #pragma unroll
        for (int r = 0; r < 8; ++r) {
            float d  = t ? bd1[r] : bd0[r];
            int   ix = t ? bi1[r] : bi0[r];
            #pragma unroll
            for (int m = 1; m <= 8; m <<= 1) {
                const float od = __shfl_xor(d, m, 32);
                const int   oi = __shfl_xor(ix, m, 32);
                if (od < d || (od == d && oi < ix)) { d = od; ix = oi; }
            }
            if (lo == 0) idx_lds[w * 32 + t * 16 + hi * 8 + r] = ix;
        }
    }
    __syncthreads();

    // ---- Gather winning codebook rows (f32), coalesced NCHW store ----
    float* ob = out + (size_t)b * 256 * 4096 + hw0;
    const float* crow = cb + (size_t)idx_lds[tid] * 256;
    for (int c = 0; c < 256; ++c) {
        ob[(size_t)c * 4096 + tid] = crow[c];
    }
}

extern "C" void kernel_launch(void* const* d_in, const int* in_sizes, int n_in,
                              void* d_out, int out_size, void* d_ws, size_t ws_size,
                              hipStream_t stream) {
    const float* x  = (const float*)d_in[0];   // [32,256,64,64] f32
    const float* cb = (const float*)d_in[1];   // [1024,256] f32
    float* out = (float*)d_out;                // [32,256,64,64] f32

    unsigned short* cbbf = (unsigned short*)d_ws;                       // 512 KB bf16 codebook
    float* esq = (float*)((char*)d_ws + (size_t)1024 * 256 * 2);        // 4 KB |e|^2

    vq_prep<<<1024, 256, 0, stream>>>(cb, cbbf, esq);
    vq_main<<<512, 256, 0, stream>>>(x, cb, cbbf, esq, out);
}
